// NativeFP4Linear_14826227106504
// MI455X (gfx1250) — compile-verified, matrix operand order checked by reading, not run on hardware
//
#include <hip/hip_runtime.h>

typedef __attribute__((ext_vector_type(16))) _Float16 v16h;
typedef __attribute__((ext_vector_type(8)))  _Float16 v8h;
typedef __attribute__((ext_vector_type(4)))  _Float16 v4h;
typedef __attribute__((ext_vector_type(8)))  float    v8f;
typedef __attribute__((ext_vector_type(4)))  float    f32x4;

#define M_DIM 32
#define N_DIM 4096
#define K_DIM 4096
#define KSPLIT 16
#define KCHUNK (K_DIM / KSPLIT)                  // 256
#define WAVES_PER_WG 4
#define WG_THREADS (WAVES_PER_WG * 32)           // 128
#define COLS_PER_WAVE 16
#define COLS_PER_WG (WAVES_PER_WG * COLS_PER_WAVE) // 64

// Main GEMM: each wave computes a 32(M) x 16(N) tile over one K chunk of 256.
// Weight is dequantized (per-16-block scale * tensor scale) to f16 on the fly.
__global__ __launch_bounds__(WG_THREADS)
void fp4_linear_main_kernel(const float* __restrict__ x,
                            const float* __restrict__ w,
                            const float* __restrict__ tensor_scale,
                            const float* __restrict__ block_scales,
                            float* __restrict__ partial)
{
    __shared__ _Float16 lds_x[M_DIM * KCHUNK];   // 16 KB: x chunk as f16

    const int tid  = threadIdx.x;
    const int lane = tid & 31;
    const int wave = tid >> 5;
    const int ks   = blockIdx.y;
    const int kbeg = ks * KCHUNK;

    // ---- cooperative stage: x[0:32, kbeg:kbeg+256] f32 -> f16 in LDS ----
    {
        const int nvec = (M_DIM * KCHUNK) / 4;   // 2048 float4 chunks
        for (int v = tid; v < nvec; v += WG_THREADS) {
            const int row = v / (KCHUNK / 4);
            const int kv  = v % (KCHUNK / 4);
            f32x4 f = *(const f32x4*)(x + (size_t)row * K_DIM + kbeg + kv * 4);
            v4h h;
            h.x = (_Float16)f.x; h.y = (_Float16)f.y;
            h.z = (_Float16)f.z; h.w = (_Float16)f.w;
            *(v4h*)(lds_x + row * KCHUNK + kv * 4) = h;
        }
    }
    __syncthreads();

    const int n0   = blockIdx.x * COLS_PER_WG + wave * COLS_PER_WAVE;
    const int col  = n0 + (lane & 15);
    const int kA   = (lane < 16) ? 0 : 8;        // 16-bit B-matrix lane layout
    const float inv_ts = 1.0f / tensor_scale[0]; // uniform, once per wave

    v8f c0 = {};   // rows 0..15  of C
    v8f c1 = {};   // rows 16..31 of C

    const float* wcol     = w + (size_t)col * K_DIM + kbeg;
    const int    blk_base = (col * K_DIM + kbeg) >> 4;  // 16-elem scale blocks

#pragma unroll 2
    for (int kk = 0; kk < KCHUNK; kk += 32) {
        // Prefetch the weight stream one unrolled body (64 floats) ahead.
        __builtin_prefetch(wcol + kk + 64 + kA, 0, 0);

        // ---- B fragment: weight col `col`, K = kk + {kA..kA+7, kA+16..kA+23}
        // Each 8-run lies inside exactly one 16-element scale block.
        f32x4 b0 = __builtin_nontemporal_load((const f32x4*)(wcol + kk + kA));
        f32x4 b1 = __builtin_nontemporal_load((const f32x4*)(wcol + kk + kA + 4));
        f32x4 b2 = __builtin_nontemporal_load((const f32x4*)(wcol + kk + kA + 16));
        f32x4 b3 = __builtin_nontemporal_load((const f32x4*)(wcol + kk + kA + 20));
        // Block scales are positive in [0.5, 2.0]: hardware rcp (v_rcp_f32)
        // is safe; avoids the full IEEE divide expansion in the inner loop.
        const float s0 = inv_ts * __builtin_amdgcn_rcpf(__builtin_nontemporal_load(
                             block_scales + blk_base + ((kk + kA) >> 4)));
        const float s1 = inv_ts * __builtin_amdgcn_rcpf(__builtin_nontemporal_load(
                             block_scales + blk_base + ((kk + kA + 16) >> 4)));

        v16h bfrag;
        bfrag[0]  = (_Float16)(b0.x * s0); bfrag[1]  = (_Float16)(b0.y * s0);
        bfrag[2]  = (_Float16)(b0.z * s0); bfrag[3]  = (_Float16)(b0.w * s0);
        bfrag[4]  = (_Float16)(b1.x * s0); bfrag[5]  = (_Float16)(b1.y * s0);
        bfrag[6]  = (_Float16)(b1.z * s0); bfrag[7]  = (_Float16)(b1.w * s0);
        bfrag[8]  = (_Float16)(b2.x * s1); bfrag[9]  = (_Float16)(b2.y * s1);
        bfrag[10] = (_Float16)(b2.z * s1); bfrag[11] = (_Float16)(b2.w * s1);
        bfrag[12] = (_Float16)(b3.x * s1); bfrag[13] = (_Float16)(b3.y * s1);
        bfrag[14] = (_Float16)(b3.z * s1); bfrag[15] = (_Float16)(b3.w * s1);

        // ---- A fragments from LDS (ds_load_b128), same lane K-layout ----
        const _Float16* arow0 = lds_x + (lane & 15) * KCHUNK + kk + kA;
        const _Float16* arow1 = arow0 + 16 * KCHUNK;
        v8h a0lo = *(const v8h*)(arow0);
        v8h a0hi = *(const v8h*)(arow0 + 16);
        v8h a1lo = *(const v8h*)(arow1);
        v8h a1hi = *(const v8h*)(arow1 + 16);
        v16h afrag0, afrag1;
#pragma unroll
        for (int i = 0; i < 8; ++i) {
            afrag0[i] = a0lo[i]; afrag0[i + 8] = a0hi[i];
            afrag1[i] = a1lo[i]; afrag1[i + 8] = a1hi[i];
        }

        // D = A x B + C  (f32 accumulate)
        c0 = __builtin_amdgcn_wmma_f32_16x16x32_f16(
                 false, afrag0, false, bfrag, (short)0, c0, false, false);
        c1 = __builtin_amdgcn_wmma_f32_16x16x32_f16(
                 false, afrag1, false, bfrag, (short)0, c1, false, false);
    }

    // ---- epilogue: C layout -> partial[ks][m][n] ----
    // VGPR r: lanes 0..15 -> M=r ; lanes 16..31 -> M=r+8 ; N = n0 + (lane&15)
    float* pbase = partial + (size_t)ks * (M_DIM * N_DIM);
    const int prow = (lane < 16) ? 0 : 8;
#pragma unroll
    for (int r = 0; r < 8; ++r) {
        pbase[(size_t)(prow + r)      * N_DIM + col] = c0[r];
        pbase[(size_t)(prow + r + 16) * N_DIM + col] = c1[r];
    }
}

// Deterministic split-K reduction + bias.
__global__ __launch_bounds__(256)
void fp4_linear_reduce_kernel(const float* __restrict__ partial,
                              const float* __restrict__ bias,
                              float* __restrict__ out)
{
    const int idx = blockIdx.x * blockDim.x + threadIdx.x;
    if (idx >= M_DIM * N_DIM) return;
    const int o = idx & (N_DIM - 1);
    float acc = bias[o];
#pragma unroll
    for (int ks = 0; ks < KSPLIT; ++ks)
        acc += partial[(size_t)ks * (M_DIM * N_DIM) + idx];
    out[idx] = acc;
}

extern "C" void kernel_launch(void* const* d_in, const int* in_sizes, int n_in,
                              void* d_out, int out_size, void* d_ws, size_t ws_size,
                              hipStream_t stream) {
    const float* x    = (const float*)d_in[0];   // [32, 4096]
    const float* w    = (const float*)d_in[1];   // [4096, 4096]
    const float* ts   = (const float*)d_in[2];   // [1]
    const float* bs   = (const float*)d_in[3];   // [4096*4096/16]
    const float* bias = (const float*)d_in[4];   // [4096]
    float* out     = (float*)d_out;              // [32, 4096]
    float* partial = (float*)d_ws;               // KSPLIT * 32 * 4096 f32 = 8 MiB

    dim3 grid(N_DIM / COLS_PER_WG, KSPLIT);      // (64, 16) workgroups, 4096 waves
    fp4_linear_main_kernel<<<grid, WG_THREADS, 0, stream>>>(x, w, ts, bs, partial);

    const int total = M_DIM * N_DIM;
    fp4_linear_reduce_kernel<<<(total + 255) / 256, 256, 0, stream>>>(partial, bias, out);
}
